// MultiheadMaskedAttention_52055003628067
// MI455X (gfx1250) — matrix operand order repaired and checked by reading, not compiled, not run on hardware
//
#include <hip/hip_runtime.h>

// ---------------------------------------------------------------------------
// MI455X (gfx1250) fused multihead causal attention
//   B=16, S=1024, D=768, H=12, HS=64
// bf16 WMMA (v_wmma_f32_16x16x32_bf16) everywhere, f32 accumulation,
// flash attention so the ~800MB score matrix never touches HBM, and
// GLOBAL_LOAD_ASYNC_TO_LDS_B128 staging where the tile needs no transpose.
// ---------------------------------------------------------------------------

typedef __bf16 bf16_t;
typedef __attribute__((ext_vector_type(16))) __bf16 v16bf;
typedef __attribute__((ext_vector_type(8)))  float  v8f;

#define AB 16      // batch
#define AS 1024    // sequence
#define AD 768     // model dim
#define AH 12      // heads
#define AHS 64     // head size
#define AD3 2304   // 3*D

// ---------------------------------------------------------------------------
// Async global->LDS (gfx1250): guarded so missing builtins fall back cleanly.
// Builtin params are pointers to 16-byte int vectors (per hipcc diagnostic).
// ---------------------------------------------------------------------------
#if defined(__gfx1250__) &&                                                    \
    __has_builtin(__builtin_amdgcn_global_load_async_to_lds_b128) &&           \
    __has_builtin(__builtin_amdgcn_s_wait_asynccnt)
#define USE_ASYNC_LDS 1
#else
#define USE_ASYNC_LDS 0
#endif

#if USE_ASYNC_LDS
typedef int v4i_t __attribute__((vector_size(16)));
typedef __attribute__((address_space(1))) v4i_t as1_v4i;
typedef __attribute__((address_space(3))) v4i_t as3_v4i;
// Generic LDS pointers carry the LDS byte offset in the low 32 bits (ISA
// aperture rule), so inttoptr to AS(3) matches the addrspacecast result.
__device__ __forceinline__ void async_copy16(const void* g, void* l) {
  __builtin_amdgcn_global_load_async_to_lds_b128(
      (as1_v4i*)(uintptr_t)g, (as3_v4i*)(uintptr_t)l, 0, 0);
}
#endif

// ---------------------------------------------------------------------------
// WMMA helpers
// ---------------------------------------------------------------------------
__device__ __forceinline__ v8f wmma_bf16(v16bf a, v16bf b, v8f c) {
  // (neg_a, A, neg_b, B, c_mod, C, reuse_a, reuse_b)
  return __builtin_amdgcn_wmma_f32_16x16x32_bf16(false, a, false, b,
                                                 (short)0, c, false, false);
}

// A-matrix fragment (16x32 bf16, MxK), gathered from row-major LDS tile.
// Per ISA: lane m=L%16, half=L/16; VGPR i holds K pair
//   k = (i/4)*16 + half*8 + (i%4)*2   (pairs contiguous -> ds_load_b128 x2)
__device__ __forceinline__ v16bf lds_load_a16(const bf16_t* base, int ldk,
                                              int row_base, int kbase) {
  const int lane = threadIdx.x & 31;
  const int m = lane & 15, half = lane >> 4;
  union { v16bf v; unsigned u[8]; } r;
  const unsigned* p = (const unsigned*)base;
  const int ro = (row_base + m) * ldk + kbase;
#pragma unroll
  for (int i = 0; i < 8; ++i) {
    const int k = ((i >> 2) << 4) | (half << 3) | ((i & 3) << 1);
    r.u[i] = p[(ro + k) >> 1];
  }
  return r.v;
}

// B-matrix fragment (32x16 bf16, KxN), gathered from a TRANSPOSED [n][k]
// LDS tile so the (k,k+1) pair is contiguous.
// Per ISA: lane n=L%16, half=L/16; VGPR i holds k = (i/4)*16 + half*8 + (i%4)*2.
__device__ __forceinline__ v16bf lds_load_b16_T(const bf16_t* baseT, int ldk,
                                                int col_base, int kbase) {
  const int lane = threadIdx.x & 31;
  const int n = lane & 15, half = lane >> 4;
  union { v16bf v; unsigned u[8]; } r;
  const unsigned* p = (const unsigned*)baseT;
  const int ro = (col_base + n) * ldk + kbase;
#pragma unroll
  for (int i = 0; i < 8; ++i) {
    const int k = ((i >> 2) << 4) | (half << 3) | ((i & 3) << 1);
    r.u[i] = p[(ro + k) >> 1];
  }
  return r.v;
}

// ---------------------------------------------------------------------------
// fp32 -> bf16 conversion
// ---------------------------------------------------------------------------
__global__ __launch_bounds__(256)
void cvt_f32_bf16(const float* __restrict__ in, bf16_t* __restrict__ out, int n) {
  int i = blockIdx.x * blockDim.x + threadIdx.x;
  const int stride = gridDim.x * blockDim.x;
  for (; i < n; i += stride) out[i] = (bf16_t)in[i];
}

// ---------------------------------------------------------------------------
// Tiled bf16 WMMA GEMM:  C[M,N] = A[M,K] (bf16) * B[K,N] (bf16) + bias
// 256 threads = 8 waves; block tile 128(M) x 128(N); K step 32.
// Wave tile 32x64: 2 A-frags x 4 B-frags = 8 WMMAs per barrier pair.
// Columns < scale_cols get *scale (folds the 1/sqrt(HS) softmax scale into Q).
// ---------------------------------------------------------------------------
template <bool OUT_BF16>
__global__ __launch_bounds__(256)
void gemm_bf16_kernel(const bf16_t* __restrict__ A, const bf16_t* __restrict__ B,
                      const float* __restrict__ bias, void* __restrict__ Cout,
                      int M, int N, int K, float scale, int scale_cols) {
  __shared__ bf16_t sA[128 * 32];   // [m][k] row-major
  __shared__ bf16_t sBT[128 * 32];  // [n][k] transposed

  const int tid   = threadIdx.x;
  const int lane  = tid & 31;
  const int wave  = tid >> 5;
  const int waveM = wave & 3;   // 4 waves along M (32 rows each)
  const int waveN = wave >> 2;  // 2 wave groups along N (64 cols each)
  const int m0 = blockIdx.x * 128;
  const int n0 = blockIdx.y * 128;

  v8f acc[2][4] = {};

  const unsigned* Bu = (const unsigned*)B;
#if !USE_ASYNC_LDS
  const unsigned* Au = (const unsigned*)A;
#endif

  for (int k0 = 0; k0 < K; k0 += 32) {
    // ---- stage A tile 128x32 bf16 (8KB = 512 x 16B chunks) ----
#if USE_ASYNC_LDS
#pragma unroll
    for (int i = 0; i < 2; ++i) {
      const int chunk = tid + i * 256;
      const int row = chunk >> 2, c = chunk & 3;  // 4 chunks per 64B row
      async_copy16((const char*)A + ((size_t)(m0 + row) * K + k0) * 2 + c * 16,
                   (char*)sA + chunk * 16);
    }
#else
#pragma unroll
    for (int i = 0; i < 8; ++i) {
      const int u = tid + i * 256;  // 2048 dwords
      const int row = u >> 4, c2 = u & 15;
      ((unsigned*)sA)[u] = Au[(((m0 + row) * K + k0) >> 1) + c2];
    }
    if (k0 + 32 < K)
      __builtin_prefetch(Au + (((m0 + (tid >> 1)) * K + k0 + 32) >> 1), 0, 1);
#endif
    // ---- stage B tile transposed (32x128 -> [n][k]; 2048 dwords) ----
#pragma unroll
    for (int i = 0; i < 8; ++i) {
      const int u = tid + i * 256;
      const int k = u >> 6, n2 = u & 63;
      union { unsigned u32; bf16_t h[2]; } w;
      w.u32 = Bu[(((k0 + k) * N + n0) >> 1) + n2];
      const int n = n2 * 2;
      sBT[n * 32 + k]       = w.h[0];
      sBT[(n + 1) * 32 + k] = w.h[1];
    }
    if (k0 + 32 < K)  // global_prefetch_b8 of next B tile
      __builtin_prefetch(Bu + (((k0 + 32 + (tid & 31)) * N + n0) >> 1), 0, 1);
#if USE_ASYNC_LDS
    __builtin_amdgcn_s_wait_asynccnt(0);
#endif
    __syncthreads();

    const v16bf a0 = lds_load_a16(sA, 32, waveM * 32, 0);
    const v16bf a1 = lds_load_a16(sA, 32, waveM * 32 + 16, 0);
#pragma unroll
    for (int t = 0; t < 4; ++t) {
      const v16bf b = lds_load_b16_T(sBT, 32, waveN * 64 + t * 16, 0);
      acc[0][t] = wmma_bf16(a0, b, acc[0][t]);
      acc[1][t] = wmma_bf16(a1, b, acc[1][t]);
    }
    __syncthreads();
  }

  const int half = lane >> 4, nl = lane & 15;
#pragma unroll
  for (int t = 0; t < 4; ++t) {
    const int col = n0 + waveN * 64 + t * 16 + nl;
    const float bc = bias[col];
    const float sc = (col < scale_cols) ? scale : 1.0f;
#pragma unroll
    for (int s = 0; s < 2; ++s) {
#pragma unroll
      for (int r = 0; r < 8; ++r) {
        const int row = m0 + waveM * 32 + s * 16 + r + 8 * half;  // m = r+8*half
        const float v = (acc[s][t][r] + bc) * sc;
        if (OUT_BF16) ((bf16_t*)Cout)[(size_t)row * N + col] = (bf16_t)v;
        else          ((float*)Cout)[(size_t)row * N + col]  = v;
      }
    }
  }
}

// ---------------------------------------------------------------------------
// Flash attention (causal, softmax scale pre-folded into Q).
// grid = (S/64, H, B); block = 128 (4 waves). Wave w owns Q rows
// [q0+16w, q0+16w+16). Streams 32-key K/V tiles; online softmax; P goes
// through per-wave LDS to re-layout C(f32) -> A(bf16) for the PV WMMAs.
// ---------------------------------------------------------------------------
__global__ __launch_bounds__(128)
void flash_attn_kernel(const bf16_t* __restrict__ qkv, bf16_t* __restrict__ av) {
  __shared__ bf16_t sQ[64 * 64];      // [qrow][hs]
  __shared__ bf16_t sK[32 * 64];      // [key][hs]  (== B^T for QK^T)
  __shared__ bf16_t sVT[64 * 32];     // [hs][key]  (== B^T for P*V)
  __shared__ bf16_t sP[4][16 * 32];   // per-wave P tile [m][key]

  const int tid  = threadIdx.x;
  const int lane = tid & 31;
  const int wave = tid >> 5;
  const int b = blockIdx.z, h = blockIdx.y;
  const int q0 = blockIdx.x * 64;
  const long baseRow = (long)b * AS;

  const unsigned* qkvU = (const unsigned*)qkv;

  // ---- stage Q tile 64x64 bf16 (8KB) ----
#if USE_ASYNC_LDS
#pragma unroll
  for (int i = 0; i < 4; ++i) {
    const int chunk = tid + i * 128;          // 512 chunks, 8 per 128B row
    const int row = chunk >> 3, c = chunk & 7;
    async_copy16((const char*)qkv + ((baseRow + q0 + row) * AD3 + h * AHS) * 2 + c * 16,
                 (char*)sQ + chunk * 16);
  }
  __builtin_amdgcn_s_wait_asynccnt(0);
#else
#pragma unroll
  for (int i = 0; i < 16; ++i) {
    const int u = tid + i * 128;
    const int row = u >> 5, c2 = u & 31;
    ((unsigned*)sQ)[u] = qkvU[(((baseRow + q0 + row) * AD3 + h * AHS) >> 1) + c2];
  }
#endif
  __syncthreads();
  const v16bf aQ0 = lds_load_a16(sQ, 64, wave * 16, 0);   // k(hs) in [0,32)
  const v16bf aQ1 = lds_load_a16(sQ, 64, wave * 16, 32);  // k(hs) in [32,64)

  float rm[8], rl[8];
#pragma unroll
  for (int r = 0; r < 8; ++r) { rm[r] = -3.0e38f; rl[r] = 0.0f; }
  v8f acc[4] = {};

  const int half = lane >> 4, nl = lane & 15;
  const int nkt = (q0 >> 5) + 2;  // causal: keys up to q0+63

  for (int kt = 0; kt < nkt; ++kt) {
    const int k0 = kt * 32;
    __syncthreads();  // previous iteration done with sK/sVT

    // ---- stage K tile 32x64 (async when available) ----
#if USE_ASYNC_LDS
#pragma unroll
    for (int i = 0; i < 2; ++i) {
      const int chunk = tid + i * 128;        // 256 chunks
      const int row = chunk >> 3, c = chunk & 7;
      async_copy16((const char*)qkv +
                       ((baseRow + k0 + row) * AD3 + AD + h * AHS) * 2 + c * 16,
                   (char*)sK + chunk * 16);
    }
#else
#pragma unroll
    for (int i = 0; i < 4; ++i) {
      const int u = tid + i * 128;
      const int row = u >> 5, c2 = u & 31;
      ((unsigned*)sK)[u] =
          qkvU[(((baseRow + k0 + row) * AD3 + AD + h * AHS) >> 1) + c2];
    }
#endif
    // ---- stage V tile transposed [hs][key] (manual: transpose needed) ----
#pragma unroll
    for (int i = 0; i < 8; ++i) {
      const int u = tid + i * 128;
      const int row = u >> 5, c2 = u & 31;
      union { unsigned u32; bf16_t hh[2]; } w;
      w.u32 = qkvU[(((baseRow + k0 + row) * AD3 + 2 * AD + h * AHS) >> 1) + c2];
      const int hs = c2 * 2;
      sVT[hs * 32 + row]       = w.hh[0];
      sVT[(hs + 1) * 32 + row] = w.hh[1];
    }
#if USE_ASYNC_LDS
    __builtin_amdgcn_s_wait_asynccnt(0);
#endif
    __syncthreads();

    // scores: S(16x32) = Q(16x64) * K^T(64x32), two 16-key tiles, two k-steps
    v8f c0 = {}, c1 = {};
    c0 = wmma_bf16(aQ0, lds_load_b16_T(sK, 64, 0, 0),  c0);
    c0 = wmma_bf16(aQ1, lds_load_b16_T(sK, 64, 0, 32), c0);
    c1 = wmma_bf16(aQ0, lds_load_b16_T(sK, 64, 16, 0),  c1);
    c1 = wmma_bf16(aQ1, lds_load_b16_T(sK, 64, 16, 32), c1);

    // causal mask + online softmax (rows live in 16-lane halves; xor 1/2/4/8
    // reduces across the 16 columns without crossing halves)
    const int key0 = k0 + nl, key1 = k0 + 16 + nl;
#pragma unroll
    for (int r = 0; r < 8; ++r) {
      const int m = r + 8 * half;
      const int qrow = q0 + wave * 16 + m;
      float x0 = (key0 <= qrow) ? c0[r] : -1.0e6f;  // reference uses -1e6
      float x1 = (key1 <= qrow) ? c1[r] : -1.0e6f;
      float mx = fmaxf(x0, x1);
      mx = fmaxf(mx, __shfl_xor(mx, 1));
      mx = fmaxf(mx, __shfl_xor(mx, 2));
      mx = fmaxf(mx, __shfl_xor(mx, 4));
      mx = fmaxf(mx, __shfl_xor(mx, 8));
      const float newm = fmaxf(rm[r], mx);
      const float corr = __expf(rm[r] - newm);
      const float p0 = __expf(x0 - newm);
      const float p1 = __expf(x1 - newm);
      float rs = p0 + p1;
      rs += __shfl_xor(rs, 1);
      rs += __shfl_xor(rs, 2);
      rs += __shfl_xor(rs, 4);
      rs += __shfl_xor(rs, 8);
      rl[r] = rl[r] * corr + rs;
      rm[r] = newm;
#pragma unroll
      for (int t = 0; t < 4; ++t) acc[t][r] *= corr;
      sP[wave][m * 32 + nl]      = (bf16_t)p0;   // C-layout -> row-major LDS
      sP[wave][m * 32 + 16 + nl] = (bf16_t)p1;
    }

    // PV: acc(16x64) += P(16x32) * V(32x64); same-wave LDS ops are in-order,
    // so the A-fragment gather below sees the sP writes without a barrier.
    const v16bf aP = lds_load_a16(sP[wave], 32, 0, 0);
#pragma unroll
    for (int t = 0; t < 4; ++t)
      acc[t] = wmma_bf16(aP, lds_load_b16_T(sVT, 32, t * 16, 0), acc[t]);
  }

  // normalize and store av (bf16, [B,S,D])
#pragma unroll
  for (int t = 0; t < 4; ++t) {
#pragma unroll
    for (int r = 0; r < 8; ++r) {
      const int m = r + 8 * half;
      const long row = baseRow + q0 + wave * 16 + m;
      const int col = h * AHS + t * 16 + nl;
      av[row * AD + col] = (bf16_t)(acc[t][r] / rl[r]);
    }
  }
}

// ---------------------------------------------------------------------------
// Launch: cvt -> QKV GEMM -> flash attention -> output GEMM
// Workspace layout (bytes):
//   xb    @ 0          : 16384*768*2   = 25,165,824
//   Wqkvb @ 25165824   : 768*2304*2    =  3,538,944
//   Wob   @ 28704768   : 768*768*2     =  1,179,648
//   qkvb  @ 29884416   : 16384*2304*2  = 75,497,472
//   avb   @ 105381888  : 16384*768*2   = 25,165,824   (total ~124.5 MB)
// ---------------------------------------------------------------------------
extern "C" void kernel_launch(void* const* d_in, const int* in_sizes, int n_in,
                              void* d_out, int out_size, void* d_ws, size_t ws_size,
                              hipStream_t stream) {
  const float* x    = (const float*)d_in[0];
  const float* Wqkv = (const float*)d_in[1];
  const float* bqkv = (const float*)d_in[2];
  const float* Wo   = (const float*)d_in[3];
  const float* bo   = (const float*)d_in[4];
  float* out = (float*)d_out;

  char* ws = (char*)d_ws;
  bf16_t* xb    = (bf16_t*)(ws);
  bf16_t* Wqkvb = (bf16_t*)(ws + 25165824);
  bf16_t* Wob   = (bf16_t*)(ws + 28704768);
  bf16_t* qkvb  = (bf16_t*)(ws + 29884416);
  bf16_t* avb   = (bf16_t*)(ws + 105381888);

  const int M = AB * AS;  // 16384

  cvt_f32_bf16<<<4096, 256, 0, stream>>>(x, xb, M * AD);
  cvt_f32_bf16<<<2048, 256, 0, stream>>>(Wqkv, Wqkvb, AD * AD3);
  cvt_f32_bf16<<<1024, 256, 0, stream>>>(Wo, Wob, AD * AD);

  // qkv = x @ W_qkv + b_qkv ; Q columns (< 768) pre-scaled by 1/sqrt(64)
  dim3 g1(M / 128, AD3 / 128);
  gemm_bf16_kernel<true><<<g1, 256, 0, stream>>>(xb, Wqkvb, bqkv, qkvb,
                                                 M, AD3, AD, 0.125f, AD);

  dim3 g2(AS / 64, AH, AB);
  flash_attn_kernel<<<g2, 128, 0, stream>>>(qkvb, avb);

  // out = av @ W_o + b_o (fp32 output)
  dim3 g3(M / 128, AD / 128);
  gemm_bf16_kernel<false><<<g3, 256, 0, stream>>>(avb, Wob, bo, out,
                                                  M, AD, AD, 1.0f, 0);
}
